// GT_block_74380243632837
// MI455X (gfx1250) — matrix operand order repaired and checked by reading, not compiled, not run on hardware
//
#include <hip/hip_runtime.h>
#include <hip/hip_bf16.h>

// ---------------- problem constants (match reference) ----------------
#define N_NODES 16384
#define NGRAPH  32
#define NPG     512
#define E_EDGES 262144
#define HDIM    64
#define HEADS   8
#define DH      8
#define TOT_EDGES (E_EDGES + N_NODES)   // incl. self loops

typedef __attribute__((ext_vector_type(8)))  float  v8f;
typedef __attribute__((ext_vector_type(16))) __bf16 v16bf;

struct U16x2 { uint4 a, b; };   // 32B container for one bf16 WMMA operand

__device__ __forceinline__ v16bf frag_ld(const __bf16* p0, const __bf16* p1)
{
    U16x2 t;
    t.a = *(const uint4*)p0;    // ds_load_b128
    t.b = *(const uint4*)p1;    // ds_load_b128
    return __builtin_bit_cast(v16bf, t);
}

__device__ __forceinline__ v16bf frag_pair(uint4 lo, uint4 hi)
{
    U16x2 t; t.a = lo; t.b = hi;
    return __builtin_bit_cast(v16bf, t);
}

union BPack { __bf16 h[2]; unsigned u; };
__device__ __forceinline__ unsigned pack2(float x, float y)
{
    BPack p; p.h[0] = (__bf16)x; p.h[1] = (__bf16)y;
    return p.u;
}

// =====================================================================
// C[16384,64] = A[16384,64] @ W[64,64] (+bias)  via bf16 WMMA.
// 128 rows/block, 8 waves; wave = 16 rows x 4 col-tiles, 2 K-steps.
// W is staged TRANSPOSED in LDS so B-fragments are contiguous b128 loads.
// OMODE: 0 = f32 out, 1 = bf16 out.
// =====================================================================
template <int OMODE, bool HAS_BIAS>
__global__ __launch_bounds__(256) void k_gemm64(
    const float* __restrict__ A, const float* __restrict__ W,
    const float* __restrict__ bias, void* __restrict__ out)
{
    __shared__ __align__(16) __bf16 As[128][72];   // row stride 144B (16B mult)
    __shared__ __align__(16) __bf16 Wt[64][72];    // Wt[n][k] = W[k][n]
    const int rowbase = blockIdx.x * 128;
    const int t = threadIdx.x;

    // stage A tile (float4 loads, packed 32-bit LDS stores)
    for (int i = 0; i < 8; ++i) {                   // 2048 float4 / 256 thr
        int idx = t + i * 256;
        int r = idx >> 4, c4 = idx & 15;
        float4 f = ((const float4*)A)[(size_t)(rowbase + r) * 16 + c4];
        *(unsigned*)&As[r][c4 * 4 + 0] = pack2(f.x, f.y);
        *(unsigned*)&As[r][c4 * 4 + 2] = pack2(f.z, f.w);
    }
    // stage W transposed
    for (int i = 0; i < 4; ++i) {                   // 1024 float4 / 256 thr
        int idx = t + i * 256;
        int r = idx >> 4, c4 = idx & 15;
        float4 f = ((const float4*)W)[r * 16 + c4];
        Wt[c4 * 4 + 0][r] = (__bf16)f.x;
        Wt[c4 * 4 + 1][r] = (__bf16)f.y;
        Wt[c4 * 4 + 2][r] = (__bf16)f.z;
        Wt[c4 * 4 + 3][r] = (__bf16)f.w;
    }
    __syncthreads();

    const int wave = t >> 5, lane = t & 31;
    const int m = lane & 15, hi = lane >> 4;
    const int rw = wave * 16;

    // A-fragments reused across all 4 column tiles
    v16bf af[2];
    for (int s = 0; s < 2; ++s)
        af[s] = frag_ld(&As[rw + m][s * 32 + hi * 8],
                        &As[rw + m][s * 32 + hi * 8 + 16]);

    for (int ct = 0; ct < 4; ++ct) {
        const int n = ct * 16 + m;
        v8f acc = {};
        for (int s = 0; s < 2; ++s) {
            v16bf bf = frag_ld(&Wt[n][s * 32 + hi * 8],
                               &Wt[n][s * 32 + hi * 8 + 16]);
            acc = __builtin_amdgcn_wmma_f32_16x16x32_bf16(
                false, af[s], false, bf, (short)0, acc, false, false);
        }
        float bv = 0.0f;
        if constexpr (HAS_BIAS) bv = bias[n];
        for (int v = 0; v < 8; ++v) {
            int row = rowbase + rw + v + hi * 8;
            float val = acc[v] + bv;
            if constexpr (OMODE == 0)
                ((float*)out)[(size_t)row * HDIM + n] = val;
            else
                ((__bf16*)out)[(size_t)row * HDIM + n] = (__bf16)val;
        }
    }
}

// =====================================================================
// es[n] = h[n].att_src ; ed[n] = h[n].att_dst   (one row per wave)
// =====================================================================
__global__ __launch_bounds__(256) void k_dots(
    const float* __restrict__ h, const float* __restrict__ as,
    const float* __restrict__ ad, float* __restrict__ es, float* __restrict__ ed)
{
    const int wave = threadIdx.x >> 5, lane = threadIdx.x & 31;
    const int row = blockIdx.x * 8 + wave;
    float h0 = h[(size_t)row * HDIM + lane];
    float h1 = h[(size_t)row * HDIM + lane + 32];
    float s = h0 * as[lane] + h1 * as[lane + 32];
    float d = h0 * ad[lane] + h1 * ad[lane + 32];
    for (int sh = 16; sh; sh >>= 1) {
        s += __shfl_xor(s, sh, 32);
        d += __shfl_xor(d, sh, 32);
    }
    if (lane == 0) { es[row] = s; ed[row] = d; }
}

// ---------------- zero fill ----------------
__global__ void k_zero(float* p, int n)
{
    int i = blockIdx.x * blockDim.x + threadIdx.x;
    if (i < n) p[i] = 0.0f;
}

// ---------------- segment softmax over edges ----------------
__device__ __forceinline__ unsigned enc_f32(float f)
{
    unsigned u = __float_as_uint(f);
    return (u & 0x80000000u) ? ~u : (u | 0x80000000u);
}
__device__ __forceinline__ float dec_f32(unsigned um)
{
    unsigned u = (um & 0x80000000u) ? (um & 0x7fffffffu) : ~um;
    return __uint_as_float(u);
}

__global__ void k_edge_max(const int* __restrict__ src, const int* __restrict__ dst,
                           const float* __restrict__ es, const float* __restrict__ ed,
                           float* __restrict__ ev, unsigned* __restrict__ mmax)
{
    int i = blockIdx.x * blockDim.x + threadIdx.x;
    if (i >= TOT_EDGES) return;
    int s, d;
    if (i < E_EDGES) { s = src[i]; d = dst[i]; } else { s = i - E_EDGES; d = s; }
    float e = es[s] + ed[d];
    e = (e > 0.0f) ? e : 0.2f * e;       // leaky_relu 0.2
    ev[i] = e;
    atomicMax(mmax + d, enc_f32(e));
}

__global__ void k_edge_sum(const int* __restrict__ src, const int* __restrict__ dst,
                           float* __restrict__ ev, const unsigned* __restrict__ mmax,
                           float* __restrict__ ssum)
{
    int i = blockIdx.x * blockDim.x + threadIdx.x;
    if (i >= TOT_EDGES) return;
    int d = (i < E_EDGES) ? dst[i] : (i - E_EDGES);
    float p = __expf(ev[i] - dec_f32(mmax[d]));
    ev[i] = p;
    atomicAdd(ssum + d, p);
}

__global__ void k_edge_scatter(const int* __restrict__ src, const int* __restrict__ dst,
                               const float* __restrict__ ev, const float* __restrict__ ssum,
                               const float* __restrict__ h, float* __restrict__ gat)
{
    long long tid = (long long)blockIdx.x * blockDim.x + threadIdx.x;
    if (tid >= (long long)TOT_EDGES * 16) return;
    int i = (int)(tid >> 4), c4 = (int)(tid & 15);
    int s, d;
    if (i < E_EDGES) { s = src[i]; d = dst[i]; } else { s = i - E_EDGES; d = s; }
    float alpha = ev[i] / (ssum[d] + 1e-16f);
    float4 hv = ((const float4*)h)[(size_t)s * 16 + c4];
    float* dp = gat + (size_t)d * HDIM + c4 * 4;
    atomicAdd(dp + 0, alpha * hv.x);
    atomicAdd(dp + 1, alpha * hv.y);
    atomicAdd(dp + 2, alpha * hv.z);
    atomicAdd(dp + 3, alpha * hv.w);
}

// =====================================================================
// out = LayerNorm( xin + relu(add + pre_bias) ) * g + b    (row/wave)
// =====================================================================
template <bool HAS_PB>
__global__ __launch_bounds__(256) void k_addln(
    const float* __restrict__ xin, const float* __restrict__ add,
    const float* __restrict__ pb, const float* __restrict__ g,
    const float* __restrict__ b, float* __restrict__ out)
{
    const int wave = threadIdx.x >> 5, lane = threadIdx.x & 31;
    const int row = blockIdx.x * 8 + wave;
    const int c0 = lane, c1 = lane + 32;
    float a0 = add[(size_t)row * HDIM + c0];
    float a1 = add[(size_t)row * HDIM + c1];
    if constexpr (HAS_PB) { a0 += pb[c0]; a1 += pb[c1]; }
    a0 = fmaxf(a0, 0.0f); a1 = fmaxf(a1, 0.0f);
    float t0 = xin[(size_t)row * HDIM + c0] + a0;
    float t1 = xin[(size_t)row * HDIM + c1] + a1;
    float s = t0 + t1, s2 = t0 * t0 + t1 * t1;
    for (int sh = 16; sh; sh >>= 1) {
        s  += __shfl_xor(s,  sh, 32);
        s2 += __shfl_xor(s2, sh, 32);
    }
    float mu  = s * (1.0f / 64.0f);
    float var = s2 * (1.0f / 64.0f) - mu * mu;
    float inv = rsqrtf(var + 1e-5f);
    out[(size_t)row * HDIM + c0] = (t0 - mu) * inv * g[c0] + b[c0];
    out[(size_t)row * HDIM + c1] = (t1 - mu) * inv * g[c1] + b[c1];
}

// =====================================================================
// Dense per-graph MHA: one WG per (graph, head, 32-row block).
// Phase A: S = (Q Kt)/sqrt(8), bf16 WMMA, K padded 8->32 (cndmask zeros).
// Phase B: row softmax on the resident 32x512 f32 panel.
// Phase C: O = P @ V, 16 K-steps; V staged transposed for b128 B-frags.
// LDS ~125 KB of CDNA5's 320 KB/WGP -> no flash-style rescaling needed.
// =====================================================================
__global__ __launch_bounds__(128) void k_attn(
    const __bf16* __restrict__ q, const __bf16* __restrict__ k,
    const __bf16* __restrict__ v, float* __restrict__ o)
{
    __shared__ __align__(16) float  S [32][516];
    __shared__ __align__(16) __bf16 P [32][520];   // row stride 1040B (16B mult)
    __shared__ __align__(16) __bf16 Ks[512][8];
    __shared__ __align__(16) __bf16 Vt[16][520];   // Vt[n][kk] = V[kk][n]; rows 8..15 zero
    __shared__ __align__(16) __bf16 Qs[32][8];

    const int bid = blockIdx.x;
    const int rb = bid & 15;            // 16 row-blocks of 32
    const int hd = (bid >> 4) & 7;      // head
    const int g  = bid >> 7;            // graph
    const int nodebase = g * NPG;

    const int t = threadIdx.x;
    const int wave = t >> 5, lane = t & 31;
    const int m = lane & 15, hi = lane >> 4;
    const uint4 z{0u, 0u, 0u, 0u};

    // stage K rows, V transposed, Q block
    for (int i = 0; i < 4; ++i) {
        int r = t + i * 128;
        *(uint4*)&Ks[r][0] = *(const uint4*)(k + (size_t)(nodebase + r) * HDIM + hd * DH);
        U16x2 vr;
        vr.a = *(const uint4*)(v + (size_t)(nodebase + r) * HDIM + hd * DH);
        __bf16 ve[8];
        *(U16x2*)ve = U16x2{vr.a, z};
        for (int j = 0; j < 8; ++j) Vt[j][r] = ve[j];
    }
    for (int idx = t; idx < 8 * 65; idx += 128) {   // zero Vt rows 8..15
        int r = 8 + idx / 65, c = (idx % 65) * 8;
        *(uint4*)&Vt[r][c] = z;
    }
    if (t < 32)
        *(uint4*)&Qs[t][0] = *(const uint4*)(q + (size_t)(nodebase + rb * 32 + t) * HDIM + hd * DH);
    __syncthreads();

    // -------- phase A: scores (rt fixed per wave; A-frag hoisted) --------
    const float scale = 0.35355339059327373f;   // 1/sqrt(DH)
    {
        const int rt = wave >> 1;
        uint4 qv = *(const uint4*)&Qs[rt * 16 + m][0];
        v16bf af = frag_pair(hi ? z : qv, z);      // K rows 8..31 are zero
        for (int i = 0; i < 16; ++i) {
            int ct = (wave & 1) * 16 + i;
            uint4 kv = *(const uint4*)&Ks[ct * 16 + m][0];
            v16bf bf = frag_pair(hi ? z : kv, z);
            v8f acc = {};
            acc = __builtin_amdgcn_wmma_f32_16x16x32_bf16(
                false, af, false, bf, (short)0, acc, false, false);
            for (int vv = 0; vv < 8; ++vv)
                S[rt * 16 + vv + hi * 8][ct * 16 + m] = acc[vv] * scale;
        }
    }
    __syncthreads();

    // -------- phase B: softmax over 512 cols --------
    for (int r = wave; r < 32; r += 4) {
        float vals[16];
        float mx = -3.0e38f;
        for (int i = 0; i < 16; ++i) {
            vals[i] = S[r][lane + i * 32];
            mx = fmaxf(mx, vals[i]);
        }
        for (int sh = 16; sh; sh >>= 1) mx = fmaxf(mx, __shfl_xor(mx, sh, 32));
        float sum = 0.0f;
        for (int i = 0; i < 16; ++i) { vals[i] = __expf(vals[i] - mx); sum += vals[i]; }
        for (int sh = 16; sh; sh >>= 1) sum += __shfl_xor(sum, sh, 32);
        float inv = 1.0f / sum;
        for (int i = 0; i < 16; ++i) P[r][lane + i * 32] = (__bf16)(vals[i] * inv);
    }
    __syncthreads();

    // -------- phase C: O = P @ V --------
    if (wave < 2) {
        int rt = wave;
        v8f acc = {};
        for (int ks = 0; ks < 16; ++ks) {
            v16bf af = frag_ld(&P[rt * 16 + m][ks * 32 + hi * 8],
                               &P[rt * 16 + m][ks * 32 + hi * 8 + 16]);
            v16bf bf = frag_ld(&Vt[m][ks * 32 + hi * 8],
                               &Vt[m][ks * 32 + hi * 8 + 16]);
            acc = __builtin_amdgcn_wmma_f32_16x16x32_bf16(
                false, af, false, bf, (short)0, acc, false, false);
        }
        if (m < DH) {
            for (int vv = 0; vv < 8; ++vv) {
                int node = nodebase + rb * 32 + rt * 16 + vv + hi * 8;
                o[(size_t)node * HDIM + hd * DH + m] = acc[vv];
            }
        }
    }
}

// =====================================================================
// host-side launch
// =====================================================================
extern "C" void kernel_launch(void* const* d_in, const int* in_sizes, int n_in,
                              void* d_out, int out_size, void* d_ws, size_t ws_size,
                              hipStream_t stream)
{
    (void)in_sizes; (void)n_in; (void)out_size; (void)ws_size;
    const float* x       = (const float*)d_in[0];
    const float* w_gat   = (const float*)d_in[1];
    const float* att_src = (const float*)d_in[2];
    const float* att_dst = (const float*)d_in[3];
    const float* b_gat   = (const float*)d_in[4];
    const float* g1      = (const float*)d_in[5];
    const float* b1      = (const float*)d_in[6];
    const float* wq      = (const float*)d_in[7];
    const float* bq      = (const float*)d_in[8];
    const float* wk      = (const float*)d_in[9];
    const float* bk      = (const float*)d_in[10];
    const float* wv      = (const float*)d_in[11];
    const float* bv      = (const float*)d_in[12];
    const float* wo      = (const float*)d_in[13];
    const float* bo      = (const float*)d_in[14];
    const float* g2      = (const float*)d_in[15];
    const float* b2      = (const float*)d_in[16];
    const int*   ei      = (const int*)d_in[17];
    const int* srcp = ei;
    const int* dstp = ei + E_EDGES;

    // workspace layout (bytes, 256-aligned)
    char* ws = (char*)d_ws;
    float*    h    = (float*)   (ws + 0);           // 4 MB
    float*    es   = (float*)   (ws + 4194304);
    float*    ed   = (float*)   (ws + 4259840);
    unsigned* mmax = (unsigned*)(ws + 4325376);
    float*    ssum = (float*)   (ws + 4390912);
    float*    ev   = (float*)   (ws + 4456448);     // (E+N) f32
    float*    gat  = (float*)   (ws + 5570560);     // 4 MB
    float*    x1   = (float*)   (ws + 9764864);     // 4 MB
    __bf16*   qb   = (__bf16*)  (ws + 13959168);    // 2 MB
    __bf16*   kb   = (__bf16*)  (ws + 16056320);
    __bf16*   vb   = (__bf16*)  (ws + 18153472);
    float*    oatt = (float*)   (ws + 20250624);    // 4 MB
    float*    o2   = (float*)   (ws + 24444928);    // 4 MB
    float*    x2   = (float*)d_out;

    // zero the atomic accumulators
    k_zero<<<(N_NODES + 255) / 256, 256, 0, stream>>>((float*)mmax, N_NODES);
    k_zero<<<(N_NODES + 255) / 256, 256, 0, stream>>>(ssum, N_NODES);
    k_zero<<<(N_NODES * HDIM + 255) / 256, 256, 0, stream>>>(gat, N_NODES * HDIM);

    // GAT branch
    k_gemm64<0, false><<<N_NODES / 128, 256, 0, stream>>>(x, w_gat, nullptr, h);
    k_dots<<<N_NODES / 8, 256, 0, stream>>>(h, att_src, att_dst, es, ed);
    k_edge_max    <<<(TOT_EDGES + 255) / 256, 256, 0, stream>>>(srcp, dstp, es, ed, ev, mmax);
    k_edge_sum    <<<(TOT_EDGES + 255) / 256, 256, 0, stream>>>(srcp, dstp, ev, mmax, ssum);
    k_edge_scatter<<<(TOT_EDGES * 16 + 255) / 256, 256, 0, stream>>>(srcp, dstp, ev, ssum, h, gat);
    k_addln<true><<<N_NODES / 8, 256, 0, stream>>>(x, gat, b_gat, g1, b1, x1);

    // dense MHA branch
    k_gemm64<1, true><<<N_NODES / 128, 256, 0, stream>>>(x1, wq, bq, qb);
    k_gemm64<1, true><<<N_NODES / 128, 256, 0, stream>>>(x1, wk, bk, kb);
    k_gemm64<1, true><<<N_NODES / 128, 256, 0, stream>>>(x1, wv, bv, vb);
    k_attn<<<NGRAPH * HEADS * 16, 128, 0, stream>>>(qb, kb, vb, oatt);
    k_gemm64<0, true><<<N_NODES / 128, 256, 0, stream>>>(oatt, wo, bo, o2);
    k_addln<false><<<N_NODES / 8, 256, 0, stream>>>(x1, o2, nullptr, g2, b2, x2);
}
